// MultiHeadAttention_72164040507449
// MI455X (gfx1250) — matrix operand order repaired and checked by reading, not compiled, not run on hardware
//
#include <hip/hip_runtime.h>
#include <hip/hip_bf16.h>

typedef __attribute__((ext_vector_type(16))) __bf16 v16bf;
typedef __attribute__((ext_vector_type(8)))  float  v8f;

#define cHID 2048
#define cNH  16
#define cHD  128
#define cB   2
#define cS   2048
#define cM   (cB * cS)   // 4096 rows

// ---------------------------------------------------------------------------
// WMMA helper: D = A(16x32 bf16) * B(32x16 bf16) + C(16x16 f32)
// ---------------------------------------------------------------------------
__device__ __forceinline__ v8f wmma_bf16(v16bf a, v16bf b, v8f c) {
    return __builtin_amdgcn_wmma_f32_16x16x32_bf16(
        false, a, false, b, (short)0, c, false, false);
}

// A-fragment (16x32, row-major source, row stride ld elements).
__device__ __forceinline__ v16bf frag_A(const __bf16* __restrict__ s, int ld) {
    const int lane = threadIdx.x & 31;
    const int half = lane >> 4;
    const int m    = lane & 15;
    v16bf a;
#pragma unroll
    for (int v = 0; v < 8; ++v) {
        const int k0 = 2 * v + 16 * half;
        a[2 * v]     = s[m * ld + k0];
        a[2 * v + 1] = s[m * ld + k0 + 1];
    }
    return a;
}

// B-fragment from B^T stored row-major (rows = N, cols = K): contiguous K.
__device__ __forceinline__ v16bf frag_BT(const __bf16* __restrict__ s, int ld) {
    const int lane = threadIdx.x & 31;
    const int half = lane >> 4;
    const int n    = lane & 15;
    v16bf b;
#pragma unroll
    for (int v = 0; v < 8; ++v) {
        const int k0 = 2 * v + 16 * half;
        b[2 * v]     = s[n * ld + k0];
        b[2 * v + 1] = s[n * ld + k0 + 1];
    }
    return b;
}

// Butterfly max over 16-lane rows: pure-VALU v_permlane16_b32 path.
__device__ __forceinline__ float rowmax16(float x) {
#if __has_builtin(__builtin_amdgcn_permlane16)
    union { float f; int i; } u, v;
    u.f = x;
    v.i = __builtin_amdgcn_permlane16(u.i, u.i, 0x67452301, 0xEFCDAB89, false, false); // xor 1
    u.f = fmaxf(u.f, v.f);
    v.i = __builtin_amdgcn_permlane16(u.i, u.i, 0x54761032, 0xDCFE98BA, false, false); // xor 2
    u.f = fmaxf(u.f, v.f);
    v.i = __builtin_amdgcn_permlane16(u.i, u.i, 0x32107654, 0xBA98FEDC, false, false); // xor 4
    u.f = fmaxf(u.f, v.f);
    v.i = __builtin_amdgcn_permlane16(u.i, u.i, 0xFEDCBA98, 0x76543210, false, false); // xor 8
    u.f = fmaxf(u.f, v.f);
    return u.f;
#else
#pragma unroll
    for (int off = 1; off < 16; off <<= 1) x = fmaxf(x, __shfl_xor(x, off, 16));
    return x;
#endif
}

// ---------------------------------------------------------------------------
// Elementwise f32 -> bf16 convert (one-time; memory-bound)
// ---------------------------------------------------------------------------
__global__ __launch_bounds__(256) void cvt_bf16(const float* __restrict__ s,
                                                __bf16* __restrict__ d, int n4) {
    const int i = blockIdx.x * 256 + threadIdx.x;
    if (i < n4) {
        const float4 f = reinterpret_cast<const float4*>(s)[i];
        __bf16* p = d + (size_t)i * 4;
        p[0] = (__bf16)f.x; p[1] = (__bf16)f.y;
        p[2] = (__bf16)f.z; p[3] = (__bf16)f.w;
    }
}

// ---------------------------------------------------------------------------
// GEMM: Y[M,N] = (X[M,K] * W[N,K]^T + bias[N]) * ps     (all-bf16 inputs)
// Block tile 128x128, BK=32, 256 threads = 8 waves (4x2), wave tile 32x64.
// DOUBLE-BUFFERED LDS with CDNA5 async loads: issue next tile, then
// s_wait_asynccnt 4 retires only the older tile (in-order ASYNCcnt).
// OMODE: 0 = f32 row-major, 1 = bf16 row-major, 2 = bf16 V-transpose (b,h,d,s)
// ---------------------------------------------------------------------------
template <int OMODE>
__global__ __launch_bounds__(256) void gemm_bf16(
    const __bf16* __restrict__ X, const __bf16* __restrict__ W,
    const float* __restrict__ bias, void* __restrict__ Yv,
    int M, int N, int K, float ps) {
    constexpr int BM = 128, BN = 128, BK = 32, LD = BK + 8;
    __shared__ __bf16 Xs[2][BM * LD];
    __shared__ __bf16 Ws[2][BN * LD];

    const int tid  = threadIdx.x;
    const int wave = tid >> 5;
    const int wy   = wave >> 1;   // 0..3 : 32-row strip
    const int wx   = wave & 1;    // 0..1 : 64-col strip
    const int bm0  = blockIdx.y * BM;
    const int bn0  = blockIdx.x * BN;

    // Stage one 128x32 X-tile + 128x32 W-tile into LDS buffer `buf`.
    // 4 async-load instructions per thread per tile (2 X + 2 W).
    auto stage = [&](int k0, int buf) {
#pragma unroll
        for (int i = tid * 8; i < BM * BK; i += 256 * 8) {
            const int r = i >> 5, c = i & 31;
            const uint32_t loff  = (uint32_t)(uintptr_t)(&Xs[buf][r * LD + c]);
            const uint64_t gaddr =
                (uint64_t)(uintptr_t)(X + (size_t)(bm0 + r) * K + k0 + c);
            asm volatile("global_load_async_to_lds_b128 %0, %1, off"
                         :: "v"(loff), "v"(gaddr) : "memory");
        }
#pragma unroll
        for (int i = tid * 8; i < BN * BK; i += 256 * 8) {
            const int r = i >> 5, c = i & 31;
            const uint32_t loff  = (uint32_t)(uintptr_t)(&Ws[buf][r * LD + c]);
            const uint64_t gaddr =
                (uint64_t)(uintptr_t)(W + (size_t)(bn0 + r) * K + k0 + c);
            asm volatile("global_load_async_to_lds_b128 %0, %1, off"
                         :: "v"(loff), "v"(gaddr) : "memory");
        }
    };

    v8f acc[2][4];
#pragma unroll
    for (int i = 0; i < 2; ++i)
#pragma unroll
        for (int j = 0; j < 4; ++j) acc[i][j] = (v8f){};

    stage(0, 0);
    const int nkt = K / BK;
    for (int kt = 0; kt < nkt; ++kt) {
        const int cur = kt & 1;
        if (kt + 1 < nkt) {
            stage((kt + 1) * BK, cur ^ 1);
            // Older tile's 4 async ops retired; newer 4 may remain in flight.
            asm volatile("s_wait_asynccnt 0x4" ::: "memory");
        } else {
            asm volatile("s_wait_asynccnt 0x0" ::: "memory");
        }
        __syncthreads();

        v16bf af[2], bfr[4];
#pragma unroll
        for (int i = 0; i < 2; ++i)
            af[i] = frag_A(&Xs[cur][(wy * 32 + i * 16) * LD], LD);
#pragma unroll
        for (int j = 0; j < 4; ++j)
            bfr[j] = frag_BT(&Ws[cur][(wx * 64 + j * 16) * LD], LD);
#pragma unroll
        for (int i = 0; i < 2; ++i)
#pragma unroll
            for (int j = 0; j < 4; ++j)
                acc[i][j] = wmma_bf16(af[i], bfr[j], acc[i][j]);
        __syncthreads();
    }

    // Epilogue: C/D layout -> (m = r + 8*half, n = lane&15)
    const int lane = tid & 31;
    const int half = lane >> 4;
    const int nl   = lane & 15;
#pragma unroll
    for (int i = 0; i < 2; ++i)
#pragma unroll
        for (int j = 0; j < 4; ++j)
#pragma unroll
            for (int r = 0; r < 8; ++r) {
                const int gm = bm0 + wy * 32 + i * 16 + r + 8 * half;
                const int gn = bn0 + wx * 64 + j * 16 + nl;
                const float y = (acc[i][j][r] + bias[gn]) * ps;
                if constexpr (OMODE == 0) {
                    ((float*)Yv)[(size_t)gm * N + gn] = y;
                } else if constexpr (OMODE == 1) {
                    ((__bf16*)Yv)[(size_t)gm * N + gn] = (__bf16)y;
                } else {  // V-transpose: (b,h,d,s)
                    const int b = gm >> 11, s_ = gm & (cS - 1);
                    const int h = gn >> 7,  d_ = gn & (cHD - 1);
                    ((__bf16*)Yv)[(((size_t)b * cNH + h) * cHD + d_) * cS + s_] =
                        (__bf16)y;
                }
            }
}

// ---------------------------------------------------------------------------
// Causal flash attention. Grid (S/64, NH, B), 128 threads = 4 waves.
// Q pre-scaled by 1/sqrt(HD). V supplied transposed (b,h,d,s).
// Row-sum of P computed by an extra WMMA against an all-ones B matrix;
// row-max via v_permlane16_b32 butterfly (no LDS traffic).
// ---------------------------------------------------------------------------
__global__ __launch_bounds__(128) void flash_attn(
    const __bf16* __restrict__ Qw, const __bf16* __restrict__ Kw,
    const __bf16* __restrict__ Vt, __bf16* __restrict__ AO) {
    constexpr int LDP = 40;
    __shared__ __bf16 Ps[4 * 16 * LDP];

    const int tid  = threadIdx.x;
    const int wid  = tid >> 5;
    const int lane = tid & 31;
    const int half = lane >> 4;
    const int nl   = lane & 15;
    const int h    = blockIdx.y;
    const int b    = blockIdx.z;
    const int qbase = blockIdx.x * 64 + wid * 16;

    const size_t bh = (size_t)b * cS * cHID + (size_t)h * cHD;
    const __bf16* Qp  = Qw + bh;
    const __bf16* Kp  = Kw + bh;
    const __bf16* Vtp = Vt + ((size_t)b * cNH + h) * (size_t)cHD * cS;
    __bf16* Pw = &Ps[wid * 16 * LDP];

    v16bf ones;
#pragma unroll
    for (int i = 0; i < 16; ++i) ones[i] = (__bf16)1.0f;

    v16bf qf[4];
#pragma unroll
    for (int kk = 0; kk < 4; ++kk)
        qf[kk] = frag_A(Qp + (size_t)qbase * cHID + kk * 32, cHID);

    v8f o[8];
#pragma unroll
    for (int j = 0; j < 8; ++j) o[j] = (v8f){};
    float mrow[8], lrow[8];
#pragma unroll
    for (int r = 0; r < 8; ++r) { mrow[r] = -1e30f; lrow[r] = 0.0f; }

    const int nch = (qbase + 16 + 31) >> 5;

    for (int c = 0; c < nch; ++c) {
        const int kc0 = c * 32;

        // S = Q * K^T (scores pre-scaled via Q)
        v8f s0 = (v8f){}, s1 = (v8f){};
#pragma unroll
        for (int kk = 0; kk < 4; ++kk) {
            v16bf kf0 = frag_BT(Kp + (size_t)kc0 * cHID + kk * 32, cHID);
            s0 = wmma_bf16(qf[kk], kf0, s0);
            v16bf kf1 = frag_BT(Kp + (size_t)(kc0 + 16) * cHID + kk * 32, cHID);
            s1 = wmma_bf16(qf[kk], kf1, s1);
        }

        // Online softmax: row m = r + 8*half, col n = lane&15
        float alpha[8];
#pragma unroll
        for (int r = 0; r < 8; ++r) {
            const int m    = r + 8 * half;
            const int qrow = qbase + m;
            float v0 = s0[r];
            float v1 = s1[r];
            if (kc0 + nl > qrow)      v0 = -1e30f;
            if (kc0 + 16 + nl > qrow) v1 = -1e30f;

            const float t    = rowmax16(fmaxf(v0, v1));
            const float mnew = fmaxf(mrow[r], t);
            alpha[r] = __expf(mrow[r] - mnew);
            mrow[r]  = mnew;
            const float p0 = __expf(v0 - mnew);
            const float p1 = __expf(v1 - mnew);

            Pw[m * LDP + nl]      = (__bf16)p0;
            Pw[m * LDP + 16 + nl] = (__bf16)p1;
        }

        // Rescale running output
#pragma unroll
        for (int j = 0; j < 8; ++j)
#pragma unroll
            for (int r = 0; r < 8; ++r) o[j][r] *= alpha[r];

        // Drain same-wave LDS stores, reload P as A-fragment
        asm volatile("s_wait_dscnt 0" ::: "memory");
        v16bf pa = frag_A(Pw, LDP);

        // Row-sum of P via WMMA against ones (broadcast per-row in C layout)
        v8f rsum = wmma_bf16(pa, ones, (v8f){});
#pragma unroll
        for (int r = 0; r < 8; ++r)
            lrow[r] = fmaf(lrow[r], alpha[r], rsum[r]);

        // O += P(16x32) * V(32x128) — V^T rows give contiguous-K fragments
#pragma unroll
        for (int j = 0; j < 8; ++j) {
            v16bf vf = frag_BT(Vtp + (size_t)(j * 16) * cS + kc0, cS);
            o[j] = wmma_bf16(pa, vf, o[j]);
        }
    }

    // Normalize (reciprocal per row) and store (bf16, (B,S,NH*HD) row-major)
#pragma unroll
    for (int r = 0; r < 8; ++r) {
        const int m = r + 8 * half;
        const float il = 1.0f / lrow[r];
        __bf16* dst =
            AO + ((size_t)b * cS + qbase + m) * cHID + (size_t)h * cHD + nl;
#pragma unroll
        for (int j = 0; j < 8; ++j) dst[j * 16] = (__bf16)(o[j][r] * il);
    }
}

// ---------------------------------------------------------------------------
// Host launch
// ---------------------------------------------------------------------------
extern "C" void kernel_launch(void* const* d_in, const int* in_sizes, int n_in,
                              void* d_out, int out_size, void* d_ws, size_t ws_size,
                              hipStream_t stream) {
    const float* hs = (const float*)d_in[0];
    const float* Wq = (const float*)d_in[1];
    const float* bq = (const float*)d_in[2];
    const float* Wk = (const float*)d_in[3];
    const float* bk = (const float*)d_in[4];
    const float* Wv = (const float*)d_in[5];
    const float* bv = (const float*)d_in[6];
    const float* Wo = (const float*)d_in[7];
    const float* bo = (const float*)d_in[8];

    const size_t matMK = (size_t)cM * cHID;     // 8388608
    const size_t matNK = (size_t)cHID * cHID;   // 4194304

    __bf16* Xb  = (__bf16*)d_ws;
    __bf16* Wqb = Xb + matMK;
    __bf16* Wkb = Wqb + matNK;
    __bf16* Wvb = Wkb + matNK;
    __bf16* Wob = Wvb + matNK;
    __bf16* Qw  = Wob + matNK;
    __bf16* Kw  = Qw + matMK;
    __bf16* Vt  = Kw + matMK;
    __bf16* AO  = Vt + matMK;

    // One-time f32 -> bf16 conversions
    cvt_bf16<<<(int)(matMK / 4 / 256), 256, 0, stream>>>(hs, Xb, (int)(matMK / 4));
    cvt_bf16<<<(int)(matNK / 4 / 256), 256, 0, stream>>>(Wq, Wqb, (int)(matNK / 4));
    cvt_bf16<<<(int)(matNK / 4 / 256), 256, 0, stream>>>(Wk, Wkb, (int)(matNK / 4));
    cvt_bf16<<<(int)(matNK / 4 / 256), 256, 0, stream>>>(Wv, Wvb, (int)(matNK / 4));
    cvt_bf16<<<(int)(matNK / 4 / 256), 256, 0, stream>>>(Wo, Wob, (int)(matNK / 4));

    const float scale = 0.08838834764831845f;  // 1/sqrt(128), folded into Q
    dim3 gg(cHID / 128, cM / 128);             // (16, 32)
    gemm_bf16<1><<<gg, 256, 0, stream>>>(Xb, Wqb, bq, (void*)Qw, cM, cHID, cHID, scale);
    gemm_bf16<1><<<gg, 256, 0, stream>>>(Xb, Wkb, bk, (void*)Kw, cM, cHID, cHID, 1.0f);
    gemm_bf16<2><<<gg, 256, 0, stream>>>(Xb, Wvb, bv, (void*)Vt, cM, cHID, cHID, 1.0f);

    dim3 ga(cS / 64, cNH, cB);                 // (32, 16, 2)
    flash_attn<<<ga, 128, 0, stream>>>(Qw, Kw, Vt, AO);

    gemm_bf16<0><<<gg, 256, 0, stream>>>(AO, Wob, bo, d_out, cM, cHID, cHID, 1.0f);
}
